// GCN2_45028437131841
// MI455X (gfx1250) — matrix operand order repaired and checked by reading, compiled, not run on hardware
//
#include <hip/hip_runtime.h>
#include <cstdint>
#include <cstddef>

#define NODES  50000
#define EDGES  800000
#define EPRIME (EDGES + NODES)
#define FINDIM 500
#define CDIM   128
#define LAYERS 8
#define NCLSD  7
#define ALPHA_F 0.1f
#define NEG_SLOPE_F 0.2f

typedef __attribute__((ext_vector_type(16))) __bf16 v16bf;
typedef __attribute__((ext_vector_type(2)))  __bf16 v2bf;
typedef __attribute__((ext_vector_type(8)))  float  v8f;
typedef __attribute__((ext_vector_type(8)))  unsigned int v8u;

// ---------------- helpers ----------------

// HW packed f32->bf16 conversion (single VALU op): v_cvt_pk_bf16_f32
__device__ __forceinline__ unsigned pk_bf16(float x, float y) {
#if __has_builtin(__builtin_amdgcn_cvt_pk_bf16_f32)
  v2bf r = __builtin_amdgcn_cvt_pk_bf16_f32(x, y);
  return __builtin_bit_cast(unsigned, r);
#else
  unsigned r;
  asm("v_cvt_pk_bf16_f32 %0, %1, %2" : "=v"(r) : "v"(x), "v"(y));
  return r;
#endif
}

__device__ __forceinline__ unsigned short bf16_1(float x) {
  unsigned r = pk_bf16(x, 0.0f);
  return (unsigned short)(r & 0xFFFFu);
}

__device__ __forceinline__ void gatomic_fadd(float* p, float v) {
  // guarantee HW f32 atomic (no CAS expansion)
  asm volatile("global_atomic_add_f32 %0, %1, off" :: "v"(p), "v"(v) : "memory");
}

// ---------------- weight conversion (f32 -> transposed bf16) ----------------

// lin0_w [500 x 128] -> bt [128 x 512] (K padded, zero fill)
__global__ void k_conv_lin0(const float* __restrict__ w, unsigned short* __restrict__ bt) {
  int tid = blockIdx.x * blockDim.x + threadIdx.x;
  if (tid >= CDIM * 512) return;
  int n = tid >> 9;          // /512
  int k = tid & 511;
  float v = (k < FINDIM) ? w[k * CDIM + n] : 0.0f;
  bt[tid] = bf16_1(v);
}

// batched [L x 128 x 128] -> transposed per batch
__global__ void k_conv_batch(const float* __restrict__ w, unsigned short* __restrict__ bt) {
  int tid = blockIdx.x * blockDim.x + threadIdx.x;
  if (tid >= LAYERS * CDIM * CDIM) return;
  int b = tid >> 14;
  int r = tid & 16383;
  int n = r >> 7, k = r & 127;
  bt[tid] = bf16_1(w[b * 16384 + k * CDIM + n]);
}

// W1eff[l] = (1-beta_l)*I + beta_l*w1[l], transposed bf16
__global__ void k_w1eff(const float* __restrict__ w1, unsigned short* __restrict__ bt) {
  int tid = blockIdx.x * blockDim.x + threadIdx.x;
  if (tid >= LAYERS * CDIM * CDIM) return;
  int b = tid >> 14;
  int r = tid & 16383;
  int n = r >> 7, k = r & 127;
  float beta = __logf(0.5f / (float)(b + 1) + 1.0f);
  float v = beta * w1[b * 16384 + k * CDIM + n] + ((k == n) ? (1.0f - beta) : 0.0f);
  bt[tid] = bf16_1(v);
}

// ---------------- graph norm ----------------

__global__ void k_fill(float* __restrict__ p, float v, int n) {
  int tid = blockIdx.x * blockDim.x + threadIdx.x;
  if (tid < n) p[tid] = v;
}

__global__ void k_deg(const int* __restrict__ ei, float* __restrict__ deg) {
  int e = blockIdx.x * blockDim.x + threadIdx.x;
  if (e >= EDGES) return;
  int d = ei[EDGES + e];
  gatomic_fadd(&deg[d], 1.0f);
}

__global__ void k_dinv(const float* __restrict__ deg, float* __restrict__ dinv) {
  int i = blockIdx.x * blockDim.x + threadIdx.x;
  if (i >= NODES) return;
  float d = deg[i];
  dinv[i] = (d > 0.0f) ? __frsqrt_rn(d) : 0.0f;
}

__global__ void k_ew(const int* __restrict__ ei, const float* __restrict__ dinv,
                     float* __restrict__ ew) {
  int e = blockIdx.x * blockDim.x + threadIdx.x;
  if (e >= EDGES) return;
  ew[e] = dinv[ei[e]] * dinv[ei[EDGES + e]];
}

// ---------------- WMMA GEMM: C[M x 128] = act(A_f32[M x KREAL] @ B + bias) ----------------
// Bt: bf16 [128 x KPAD] (transposed, zero-padded). One wave = 16x16 tile; 8 waves = 128 cols.

template <int KPAD, int KREAL>
__global__ __launch_bounds__(256) void k_gemm_bf16(const float* __restrict__ A,
                                                   const unsigned short* __restrict__ Bt,
                                                   const float* __restrict__ bias,
                                                   float* __restrict__ C, int relu) {
  const int lane = threadIdx.x & 31;
  const int wave = threadIdx.x >> 5;
  const int half = lane >> 4;
  const int l15  = lane & 15;
  const int row  = blockIdx.x * 16 + l15;
  const int col  = wave * 16 + l15;

  const float* arow = A + (size_t)row * KREAL;
  const unsigned short* brow = Bt + (size_t)col * KPAD;

  v8f acc = {};

#pragma unroll
  for (int k0 = 0; k0 < KPAD; k0 += 32) {
    unsigned au[8];
    const int ka = k0 + half * 8;        // K run for VGPR 0..3 (pairs)
    const int kb = k0 + 16 + half * 8;   // K run for VGPR 4..7
    if (k0 + 32 <= KREAL) {
      float4 f0 = *(const float4*)(arow + ka);
      float4 f1 = *(const float4*)(arow + ka + 4);
      float4 f2 = *(const float4*)(arow + kb);
      float4 f3 = *(const float4*)(arow + kb + 4);
      au[0] = pk_bf16(f0.x, f0.y); au[1] = pk_bf16(f0.z, f0.w);
      au[2] = pk_bf16(f1.x, f1.y); au[3] = pk_bf16(f1.z, f1.w);
      au[4] = pk_bf16(f2.x, f2.y); au[5] = pk_bf16(f2.z, f2.w);
      au[6] = pk_bf16(f3.x, f3.y); au[7] = pk_bf16(f3.z, f3.w);
    } else {
#pragma unroll
      for (int v = 0; v < 8; ++v) {
        int k = (v < 4) ? (ka + 2 * v) : (kb + 2 * (v - 4));
        float xa = (k     < KREAL) ? arow[k]     : 0.0f;
        float xb = (k + 1 < KREAL) ? arow[k + 1] : 0.0f;
        au[v] = pk_bf16(xa, xb);
      }
    }
    const unsigned short* bq = brow + k0 + half * 8;
    uint4 b0 = *(const uint4*)bq;
    uint4 b1 = *(const uint4*)(bq + 16);
    v8u aw, bw;
    aw[0] = au[0]; aw[1] = au[1]; aw[2] = au[2]; aw[3] = au[3];
    aw[4] = au[4]; aw[5] = au[5]; aw[6] = au[6]; aw[7] = au[7];
    bw[0] = b0.x; bw[1] = b0.y; bw[2] = b0.z; bw[3] = b0.w;
    bw[4] = b1.x; bw[5] = b1.y; bw[6] = b1.z; bw[7] = b1.w;
    v16bf av = __builtin_bit_cast(v16bf, aw);
    v16bf bv = __builtin_bit_cast(v16bf, bw);
    acc = __builtin_amdgcn_wmma_f32_16x16x32_bf16(false, av, false, bv, (short)0, acc,
                                                  false, false);
  }

  float badd = bias ? bias[col] : 0.0f;
  const int rbase = blockIdx.x * 16 + half * 8;
#pragma unroll
  for (int v = 0; v < 8; ++v) {
    float x = acc[v] + badd;
    if (relu) x = fmaxf(x, 0.0f);
    C[(size_t)(rbase + v) * CDIM + col] = x;
  }
}

// ---------------- per-node attention dots ----------------

__global__ void k_att(const float* __restrict__ xl, const float* __restrict__ al,
                      const float* __restrict__ ar, float* __restrict__ sl,
                      float* __restrict__ sr) {
  int i = blockIdx.x * blockDim.x + threadIdx.x;
  if (i >= NODES) return;
  const float4* xr = (const float4*)(xl + (size_t)i * CDIM);
  const float4* a4 = (const float4*)al;
  const float4* r4 = (const float4*)ar;
  float s0 = 0.0f, s1 = 0.0f;
#pragma unroll 8
  for (int j = 0; j < CDIM / 4; ++j) {
    float4 x = xr[j], a = a4[j], r = r4[j];
    s0 += x.x * a.x + x.y * a.y + x.z * a.z + x.w * a.w;
    s1 += x.x * r.x + x.y * r.y + x.z * r.z + x.w * r.w;
  }
  sl[i] = s0;
  sr[i] = s1;
}

// ---------------- edge scatter (1 wave per edge) ----------------

__global__ __launch_bounds__(256) void k_edge(const int* __restrict__ ei,
                                              const float* __restrict__ sl,
                                              const float* __restrict__ sr,
                                              const float* __restrict__ ew,
                                              const float* __restrict__ xp,
                                              const float* __restrict__ xn,
                                              float* __restrict__ agg,
                                              float* __restrict__ sig) {
  int gw = (int)((blockIdx.x * blockDim.x + threadIdx.x) >> 5);
  int lane = threadIdx.x & 31;
  if (gw >= EDGES) return;
  int s = ei[gw];
  int d = ei[EDGES + gw];
  float g = sl[s] + sr[d];
  g = (g > 0.0f) ? g : NEG_SLOPE_F * g;
  float sv = 1.0f / (1.0f + __expf(-g));
  if (lane == 0) sig[gw] = sv;
  float w = ew[gw];
  float c1 = w * sv, c2 = w * (1.0f - sv);
  float4 p = *(const float4*)(xp + (size_t)s * CDIM + lane * 4);
  float4 q = *(const float4*)(xn + (size_t)d * CDIM + lane * 4);
  float* ap = agg + (size_t)d * CDIM + lane * 4;
  gatomic_fadd(ap + 0, c1 * p.x + c2 * q.x);
  gatomic_fadd(ap + 1, c1 * p.y + c2 * q.y);
  gatomic_fadd(ap + 2, c1 * p.z + c2 * q.z);
  gatomic_fadd(ap + 3, c1 * p.w + c2 * q.w);
}

// ---------------- self-loops + residual combine: hn written in place over agg ----------------

__global__ void k_combine(float* __restrict__ agg, const float* __restrict__ xp,
                          const float* __restrict__ xn, const float* __restrict__ sl,
                          const float* __restrict__ sr, const float* __restrict__ dinv,
                          const float* __restrict__ x0, float* __restrict__ sig) {
  int t = blockIdx.x * blockDim.x + threadIdx.x;
  if (t >= NODES * 32) return;
  int i = t >> 5, lane = t & 31;
  float g = sl[i] + sr[i];
  g = (g > 0.0f) ? g : NEG_SLOPE_F * g;
  float sv = 1.0f / (1.0f + __expf(-g));
  if (lane == 0) sig[i] = sv;
  float dv = dinv[i];
  float w = dv * dv;          // self-loop edge weight
  float c1 = w * sv, c2 = w * (1.0f - sv);
  size_t off = (size_t)i * CDIM + lane * 4;
  float4 a = *(float4*)(agg + off);
  float4 p = *(const float4*)(xp + off);
  float4 q = *(const float4*)(xn + off);
  float4 x = *(const float4*)(x0 + off);
  float4 hn;
  hn.x = (1.0f - ALPHA_F) * (a.x + c1 * p.x + c2 * q.x) + ALPHA_F * x.x;
  hn.y = (1.0f - ALPHA_F) * (a.y + c1 * p.y + c2 * q.y) + ALPHA_F * x.y;
  hn.z = (1.0f - ALPHA_F) * (a.z + c1 * p.z + c2 * q.z) + ALPHA_F * x.z;
  hn.w = (1.0f - ALPHA_F) * (a.w + c1 * p.w + c2 * q.w) + ALPHA_F * x.w;
  *(float4*)(agg + off) = hn;
}

// ---------------- final classifier + log_softmax ----------------

__global__ void k_logits(const float* __restrict__ h, const float* __restrict__ w,
                         const float* __restrict__ b, float* __restrict__ out) {
  int i = blockIdx.x * blockDim.x + threadIdx.x;
  if (i >= NODES) return;
  float acc[NCLSD];
#pragma unroll
  for (int j = 0; j < NCLSD; ++j) acc[j] = b[j];
  const float* hr = h + (size_t)i * CDIM;
  for (int k = 0; k < CDIM; ++k) {
    float hv = hr[k];
#pragma unroll
    for (int j = 0; j < NCLSD; ++j) acc[j] += hv * w[k * NCLSD + j];
  }
  float m = acc[0];
#pragma unroll
  for (int j = 1; j < NCLSD; ++j) m = fmaxf(m, acc[j]);
  float ssum = 0.0f;
#pragma unroll
  for (int j = 0; j < NCLSD; ++j) ssum += __expf(acc[j] - m);
  float lse = __logf(ssum);
#pragma unroll
  for (int j = 0; j < NCLSD; ++j) out[(size_t)i * NCLSD + j] = acc[j] - m - lse;
}

// ---------------- host orchestration ----------------

extern "C" void kernel_launch(void* const* d_in, const int* in_sizes, int n_in,
                              void* d_out, int out_size, void* d_ws, size_t ws_size,
                              hipStream_t stream) {
  const float* x      = (const float*)d_in[0];
  const int*   ei     = (const int*)d_in[1];
  const float* lin0_w = (const float*)d_in[2];
  const float* lin0_b = (const float*)d_in[3];
  const float* lin1_w = (const float*)d_in[4];
  const float* lin1_b = (const float*)d_in[5];
  const float* lin_w  = (const float*)d_in[6];
  const float* att_l  = (const float*)d_in[7];
  const float* att_r  = (const float*)d_in[8];
  const float* w_p    = (const float*)d_in[9];
  const float* w_n    = (const float*)d_in[10];
  const float* w1     = (const float*)d_in[11];
  float* out = (float*)d_out;

  char* p = (char*)d_ws;
  auto carve = [&](size_t bytes) -> void* {
    void* r = (void*)p;
    p += (bytes + 255) & ~(size_t)255;
    return r;
  };
  unsigned short* lin0t = (unsigned short*)carve((size_t)CDIM * 512 * 2);
  unsigned short* lwT   = (unsigned short*)carve((size_t)LAYERS * CDIM * CDIM * 2);
  unsigned short* wpT   = (unsigned short*)carve((size_t)LAYERS * CDIM * CDIM * 2);
  unsigned short* wnT   = (unsigned short*)carve((size_t)LAYERS * CDIM * CDIM * 2);
  unsigned short* w1T   = (unsigned short*)carve((size_t)LAYERS * CDIM * CDIM * 2);
  float* deg  = (float*)carve((size_t)NODES * 4);
  float* dinv = (float*)carve((size_t)NODES * 4);
  float* slb  = (float*)carve((size_t)NODES * 4);
  float* srb  = (float*)carve((size_t)NODES * 4);
  float* ewb  = (float*)carve((size_t)EDGES * 4);
  float* x0   = (float*)carve((size_t)NODES * CDIM * 4);
  float* hb   = (float*)carve((size_t)NODES * CDIM * 4);
  float* xlag = (float*)carve((size_t)NODES * CDIM * 4);  // xl, then agg, then hn
  float* xpb  = (float*)carve((size_t)NODES * CDIM * 4);
  float* xnb  = (float*)carve((size_t)NODES * CDIM * 4);

  const int T = 256;
  // weights -> transposed bf16
  k_conv_lin0<<<(CDIM * 512 + T - 1) / T, T, 0, stream>>>(lin0_w, lin0t);
  int wcnt = LAYERS * CDIM * CDIM;
  k_conv_batch<<<(wcnt + T - 1) / T, T, 0, stream>>>(lin_w, lwT);
  k_conv_batch<<<(wcnt + T - 1) / T, T, 0, stream>>>(w_p, wpT);
  k_conv_batch<<<(wcnt + T - 1) / T, T, 0, stream>>>(w_n, wnT);
  k_w1eff<<<(wcnt + T - 1) / T, T, 0, stream>>>(w1, w1T);

  // gcn_norm (topology constant across layers)
  k_fill<<<(NODES + T - 1) / T, T, 0, stream>>>(deg, 1.0f, NODES);  // self loops
  k_deg<<<(EDGES + T - 1) / T, T, 0, stream>>>(ei, deg);
  k_dinv<<<(NODES + T - 1) / T, T, 0, stream>>>(deg, dinv);
  k_ew<<<(EDGES + T - 1) / T, T, 0, stream>>>(ei, dinv, ewb);

  // x0 = relu(x @ lin0_w + b)
  k_gemm_bf16<512, FINDIM><<<NODES / 16, T, 0, stream>>>(x, lin0t, lin0_b, x0, 1);

  const float* hcur = x0;
  float* sigbase = out + (size_t)NODES * NCLSD;
  for (int l = 0; l < LAYERS; ++l) {
    size_t wo = (size_t)l * CDIM * CDIM;
    float* sigL = sigbase + (size_t)l * EPRIME;
    k_gemm_bf16<CDIM, CDIM><<<NODES / 16, T, 0, stream>>>(hcur, lwT + wo, nullptr, xlag, 0);
    k_att<<<(NODES + T - 1) / T, T, 0, stream>>>(xlag, att_l + l * CDIM, att_r + l * CDIM,
                                                 slb, srb);
    k_gemm_bf16<CDIM, CDIM><<<NODES / 16, T, 0, stream>>>(hcur, wpT + wo, nullptr, xpb, 0);
    k_gemm_bf16<CDIM, CDIM><<<NODES / 16, T, 0, stream>>>(hcur, wnT + wo, nullptr, xnb, 0);
    k_fill<<<(NODES * CDIM + T - 1) / T, T, 0, stream>>>(xlag, 0.0f, NODES * CDIM);
    k_edge<<<((size_t)EDGES * 32 + T - 1) / T, T, 0, stream>>>(ei, slb, srb, ewb, xpb, xnb,
                                                               xlag, sigL);
    k_combine<<<(NODES * 32 + T - 1) / T, T, 0, stream>>>(xlag, xpb, xnb, slb, srb, dinv,
                                                          x0, sigL + EDGES);
    k_gemm_bf16<CDIM, CDIM><<<NODES / 16, T, 0, stream>>>(xlag, w1T + wo, nullptr, hb, 1);
    hcur = hb;
  }

  k_logits<<<(NODES + T - 1) / T, T, 0, stream>>>(hb, lin1_w, lin1_b, out);
}
